// GCNNet_40956808135250
// MI455X (gfx1250) — matrix operand order repaired and checked by reading, not compiled, not run on hardware
//
#include <hip/hip_runtime.h>

typedef _Float16 v16h __attribute__((ext_vector_type(16)));
typedef _Float16 v8h  __attribute__((ext_vector_type(8)));
typedef float    v8f  __attribute__((ext_vector_type(8)));

#define N_NODES 100000
#define N_EDGES 1000000
#define CDIV(a, b) (((a) + (b) - 1) / (b))

// ---------------- degree / symmetric norm ----------------
__global__ void k_init_deg(float* __restrict__ deg, int n) {
  int i = blockIdx.x * blockDim.x + threadIdx.x;
  if (i < n) deg[i] = 1.0f;  // self-loop contributes 1
}

__global__ void k_count_deg(const int* __restrict__ col, float* __restrict__ deg, int nE) {
  int e = blockIdx.x * blockDim.x + threadIdx.x;
  if (e < nE) atomicAdd(&deg[col[e]], 1.0f);
}

__global__ void k_finalize_dis(float* __restrict__ deg, int n) {
  int i = blockIdx.x * blockDim.x + threadIdx.x;
  if (i < n) deg[i] = rsqrtf(deg[i]);  // deg >= 1 always (self-loops)
}

// ---------------- f32 -> padded f16 packing ----------------
__global__ void k_pack_x(const float* __restrict__ x, _Float16* __restrict__ xh,
                         int n, int K, int Kp) {
  long t = (long)blockIdx.x * blockDim.x + threadIdx.x;
  if (t >= (long)n * Kp) return;
  int i = (int)(t / Kp), k = (int)(t % Kp);
  xh[t] = (k < K) ? (_Float16)x[(long)i * K + k] : (_Float16)0.0f;
}

__global__ void k_pack_w(const float* __restrict__ w, _Float16* __restrict__ wh,
                         int K, int N, int Kp, int Np) {
  int t = blockIdx.x * blockDim.x + threadIdx.x;
  if (t >= Kp * Np) return;
  int k = t / Np, n = t % Np;
  wh[t] = (k < K && n < N) ? (_Float16)w[k * N + n] : (_Float16)0.0f;
}

// ---------------- WMMA GEMM ----------------
// Block = 256 threads = 8 waves. Block handles one 16-col N-tile and 8 M-tiles.
// B panel [KP x 16] staged transposed in LDS: BT[n][k], row stride KP+8 (bank spread).
// A: [M, KP] f16 row-major, B: [KP, NP] f16 row-major, C: [M, NP] f16.
template <int KP, int NP>
__global__ void k_wmma_gemm(const _Float16* __restrict__ A, const _Float16* __restrict__ B,
                            _Float16* __restrict__ C, int Mtiles, int Mblk) {
  constexpr int BSTR = KP + 8;  // halves; +8 keeps 16B alignment, spreads banks
  __shared__ __attribute__((aligned(16))) _Float16 BT[16 * BSTR];

  const int nt = blockIdx.x / Mblk;
  const int mb = blockIdx.x - nt * Mblk;
  const int n0 = nt * 16;

  // cooperative transposed load of B panel (coalesced 32B row-chunks from global)
  for (int t = threadIdx.x; t < KP * 16; t += 256) {
    int k = t >> 4, n = t & 15;
    BT[n * BSTR + k] = B[(size_t)k * NP + n0 + n];
  }
  __syncthreads();

  const int wave = threadIdx.x >> 5;
  const int lane = threadIdx.x & 31;
  const int mtile = mb * 8 + wave;
  if (mtile >= Mtiles) return;  // wave-uniform exit -> EXEC all-1s for WMMA

  const int hi = lane >> 4;  // half-wave select
  const int lr = lane & 15;

  const _Float16* Arow  = A + (size_t)(mtile * 16 + lr) * KP;
  const _Float16* BTrow = BT + lr * BSTR;

  v8f c = {};
#pragma unroll
  for (int k = 0; k < KP; k += 32) {
    // A fragment (16-bit A 16x32): lanes 0-15 K=[k..k+7],[k+16..k+23];
    // lanes 16-31 K=[k+8..k+15],[k+24..k+31]; row M = lane&15.
    v8h alo = *(const v8h*)(Arow + k + hi * 8);
    v8h ahi = *(const v8h*)(Arow + k + 16 + hi * 8);
    v16h a;
#pragma unroll
    for (int i = 0; i < 8; ++i) { a[i] = alo[i]; a[8 + i] = ahi[i]; }

    // B fragment (16-bit B 32x16): lane column = lr; lanes 0-15 K=[k..k+7],[k+16..k+23];
    // lanes 16-31 K=[k+8..k+15],[k+24..k+31] (same K order as A).
    v8h blo = *(const v8h*)(BTrow + k + hi * 8);
    v8h bhi = *(const v8h*)(BTrow + k + 16 + hi * 8);
    v16h b;
#pragma unroll
    for (int i = 0; i < 8; ++i) { b[i] = blo[i]; b[8 + i] = bhi[i]; }

    c = __builtin_amdgcn_wmma_f32_16x16x32_f16(false, a, false, b, (short)0, c, false, false);
  }

  // D fragment: c[r] -> D[m0 + r + 8*hi][n0 + lr]
  _Float16* Crow = C + (size_t)(mtile * 16 + hi * 8) * NP + n0 + lr;
#pragma unroll
  for (int r = 0; r < 8; ++r) Crow[(size_t)r * NP] = (_Float16)c[r];
}

// ---------------- edge scatter: agg[col] += h[row] * dis[row]*dis[col] ----------------
__global__ void k_scatter(const _Float16* __restrict__ h, int Fp, int F,
                          float* __restrict__ agg,
                          const int* __restrict__ row, const int* __restrict__ col,
                          const float* __restrict__ dis, int nE) {
  int wid  = (blockIdx.x * blockDim.x + threadIdx.x) >> 5;  // one wave per edge
  int lane = threadIdx.x & 31;
  if (wid >= nE) return;
  int r = row[wid], c = col[wid];
  float nrm = dis[r] * dis[c];
  const _Float16* hr = h + (size_t)r * Fp;
  float* ac = agg + (size_t)c * F;
  for (int f = lane; f < F; f += 32)
    atomicAdd(&ac[f], (float)hr[f] * nrm);
}

__global__ void k_scatter_f1(const _Float16* __restrict__ h, int Fp,
                             float* __restrict__ agg,
                             const int* __restrict__ row, const int* __restrict__ col,
                             const float* __restrict__ dis, int nE) {
  int e = blockIdx.x * blockDim.x + threadIdx.x;
  if (e >= nE) return;
  int r = row[e], c = col[e];
  atomicAdd(&agg[c], (float)h[(size_t)r * Fp] * dis[r] * dis[c]);
}

// ---------------- fused self-loop + bias + ReLU, write f16 in place ----------------
__global__ void k_bias_relu(const float* __restrict__ agg, _Float16* __restrict__ h,
                            int Fp, int F, const float* __restrict__ b,
                            const float* __restrict__ dis, int n) {
  long t = (long)blockIdx.x * blockDim.x + threadIdx.x;
  if (t >= (long)n * F) return;
  int i = (int)(t / F), f = (int)(t % F);
  float d = dis[i];
  long hidx = (long)i * Fp + f;
  float v = agg[t] + (float)h[hidx] * d * d + b[f];
  h[hidx] = (_Float16)fmaxf(v, 0.0f);
}

__global__ void k_final(const float* __restrict__ agg, const _Float16* __restrict__ h3, int Fp,
                        const float* __restrict__ b3, const float* __restrict__ dis,
                        float* __restrict__ out, int n) {
  int i = blockIdx.x * blockDim.x + threadIdx.x;
  if (i >= n) return;
  float d = dis[i];
  out[i] = agg[i] + (float)h3[(size_t)i * Fp] * d * d + b3[0];
}

extern "C" void kernel_launch(void* const* d_in, const int* in_sizes, int n_in,
                              void* d_out, int out_size, void* d_ws, size_t ws_size,
                              hipStream_t stream) {
  const float* x  = (const float*)d_in[0];
  const int*   ei = (const int*)d_in[1];   // int32 (JAX x64 off)
  const float* W1 = (const float*)d_in[2];
  const float* b1 = (const float*)d_in[3];
  const float* W2 = (const float*)d_in[4];
  const float* b2 = (const float*)d_in[5];
  const float* W3 = (const float*)d_in[6];
  const float* b3 = (const float*)d_in[7];
  float* out = (float*)d_out;

  const int* row = ei;             // edge_index[0]
  const int* col = ei + N_EDGES;   // edge_index[1]

  // padded dims (K padded to mult of 32, N to mult of 16 matching next K)
  constexpr int Kp1 = 64,  Np1 = 320;  // 58 -> 64, 300 -> 320
  constexpr int Kp2 = 320, Np2 = 128;  // 300 -> 320, 100 -> 128
  constexpr int Kp3 = 128, Np3 = 16;   // 100 -> 128, 1 -> 16
  const int Mtiles = N_NODES / 16;     // 6250, exact
  const int Mblk = CDIV(Mtiles, 8);    // 8 M-tiles per block

  // workspace carve-out
  char* p = (char*)d_ws;
  auto alloc = [&](size_t bytes) -> char* {
    char* r = p; p += (bytes + 255) & ~(size_t)255; return r;
  };
  float*    dis = (float*)alloc((size_t)N_NODES * 4);
  _Float16* xh  = (_Float16*)alloc((size_t)N_NODES * Kp1 * 2);
  _Float16* w1h = (_Float16*)alloc((size_t)Kp1 * Np1 * 2);
  _Float16* w2h = (_Float16*)alloc((size_t)Kp2 * Np2 * 2);
  _Float16* w3h = (_Float16*)alloc((size_t)Kp3 * Np3 * 2);
  _Float16* h1  = (_Float16*)alloc((size_t)N_NODES * Np1 * 2);
  _Float16* h2  = (_Float16*)alloc((size_t)N_NODES * Np2 * 2);
  _Float16* h3  = (_Float16*)alloc((size_t)N_NODES * Np3 * 2);
  float*    agg = (float*)alloc((size_t)N_NODES * 300 * 4);  // reused across layers

  // --- degree / dis ---
  k_init_deg<<<CDIV(N_NODES, 256), 256, 0, stream>>>(dis, N_NODES);
  k_count_deg<<<CDIV(N_EDGES, 256), 256, 0, stream>>>(col, dis, N_EDGES);
  k_finalize_dis<<<CDIV(N_NODES, 256), 256, 0, stream>>>(dis, N_NODES);

  // --- pack inputs/weights to padded f16 ---
  k_pack_x<<<CDIV((long)N_NODES * Kp1, 256), 256, 0, stream>>>(x, xh, N_NODES, 58, Kp1);
  k_pack_w<<<CDIV(Kp1 * Np1, 256), 256, 0, stream>>>(W1, w1h, 58, 300, Kp1, Np1);
  k_pack_w<<<CDIV(Kp2 * Np2, 256), 256, 0, stream>>>(W2, w2h, 300, 100, Kp2, Np2);
  k_pack_w<<<CDIV(Kp3 * Np3, 256), 256, 0, stream>>>(W3, w3h, 100, 1, Kp3, Np3);

  // --- layer 1: GEMM -> scatter -> bias/relu ---
  {
    int Ntiles = Np1 / 16;  // 20
    k_wmma_gemm<Kp1, Np1><<<Ntiles * Mblk, 256, 0, stream>>>(xh, w1h, h1, Mtiles, Mblk);
    hipMemsetAsync(agg, 0, (size_t)N_NODES * 300 * 4, stream);
    k_scatter<<<CDIV((long)N_EDGES * 32, 256), 256, 0, stream>>>(
        h1, Np1, 300, agg, row, col, dis, N_EDGES);
    k_bias_relu<<<CDIV((long)N_NODES * 300, 256), 256, 0, stream>>>(
        agg, h1, Np1, 300, b1, dis, N_NODES);
  }

  // --- layer 2 ---
  {
    int Ntiles = Np2 / 16;  // 8
    k_wmma_gemm<Kp2, Np2><<<Ntiles * Mblk, 256, 0, stream>>>(h1, w2h, h2, Mtiles, Mblk);
    hipMemsetAsync(agg, 0, (size_t)N_NODES * 100 * 4, stream);
    k_scatter<<<CDIV((long)N_EDGES * 32, 256), 256, 0, stream>>>(
        h2, Np2, 100, agg, row, col, dis, N_EDGES);
    k_bias_relu<<<CDIV((long)N_NODES * 100, 256), 256, 0, stream>>>(
        agg, h2, Np2, 100, b2, dis, N_NODES);
  }

  // --- layer 3 (no relu), write f32 output ---
  {
    k_wmma_gemm<Kp3, Np3><<<1 * Mblk, 256, 0, stream>>>(h2, w3h, h3, Mtiles, Mblk);
    hipMemsetAsync(agg, 0, (size_t)N_NODES * 4, stream);
    k_scatter_f1<<<CDIV(N_EDGES, 256), 256, 0, stream>>>(h3, Np3, agg, row, col, dis, N_EDGES);
    k_final<<<CDIV(N_NODES, 256), 256, 0, stream>>>(agg, h3, Np3, b3, dis, out, N_NODES);
  }
}
